// WindowAttention_1589137899677
// MI455X (gfx1250) — matrix operand order repaired and checked by reading, not compiled, not run on hardware
//
#include <hip/hip_runtime.h>
#include <hip/hip_bf16.h>
#include <math.h>

typedef _Float16 half_t;
typedef _Float16 v16h __attribute__((ext_vector_type(16)));
typedef _Float16 h8   __attribute__((ext_vector_type(8)));
typedef float    v8f  __attribute__((ext_vector_type(8)));

#define NTOK  49
#define HEADS 12
#define HD    32
#define CDIM  384

// ---------------------------------------------------------------- f32 -> f16
__global__ void wa_f32_to_f16(const float* __restrict__ src,
                              half_t* __restrict__ dst, int n) {
  int i = blockIdx.x * blockDim.x + threadIdx.x;
  if (i < n) dst[i] = (half_t)src[i];
}

// ------------------------------------------------- bias+mask precompute
// bm[h][i][j] (64x64 padded, f32):
//   j<49 && i<49 : bias_table[rel_index[i,j]*12+h] + mask[i,j]
//   j>=49        : -1e30  (softmax pad -> P=0)
//   i>=49, j<49  : 0      (dead rows, keeps softmax finite)
__global__ void wa_bm_precompute(const float* __restrict__ mask,
                                 const float* __restrict__ bias_table,
                                 const int* __restrict__ rel_index,
                                 float* __restrict__ bm) {
  int idx = blockIdx.x * blockDim.x + threadIdx.x;
  if (idx >= HEADS * 64 * 64) return;
  int h   = idx >> 12;
  int rem = idx & 4095;
  int i = rem >> 6;
  int j = rem & 63;
  float v;
  if (j < NTOK) {
    v = (i < NTOK)
            ? bias_table[rel_index[i * NTOK + j] * HEADS + h] + mask[i * NTOK + j]
            : 0.f;
  } else {
    v = -1e30f;
  }
  bm[idx] = v;
}

// ---------------------------------------------------------------- WMMA GEMM
// C[M,N] = A[M,K]*B[K,N] (+bias). 128x128 block tile, BK=32, 256 threads,
// double-buffered LDS, 8 waves x (32x64) -> 8x v_wmma_f32_16x16x32_f16 / step.
// EPI==0: scatter to q/k/v f16 [B,H,49,32] (q scaled). EPI==1: f32 out +bias.
template <bool AHALF, int EPI>
__global__ __launch_bounds__(256)
void wa_gemm_wmma(const void* __restrict__ Aq, const half_t* __restrict__ B,
                  const float* __restrict__ bias, int M, int N, int K,
                  half_t* __restrict__ qb, half_t* __restrict__ kb,
                  half_t* __restrict__ vb, float* __restrict__ outf,
                  float scale) {
  __shared__ alignas(32) half_t As[2][128][32];   // [buf][m][k]
  __shared__ alignas(32) half_t Bs[2][128][32];   // [buf][n][k]

  const int tid  = threadIdx.x;
  const int lane = tid & 31;
  const int wv   = tid >> 5;
  const int wm   = (wv & 3) * 32;
  const int wn   = (wv >> 2) * 64;
  const int hi   = lane >> 4;
  const int ln   = lane & 15;
  const int hi8  = hi * 8;

  const int row0 = blockIdx.y * 128;
  const int col0 = blockIdx.x * 128;

  // per-thread staging coordinates
  const int ar  = tid >> 1;            // A: 2 threads/row, 16 halves each
  const int ac  = (tid & 1) * 16;
  const int bkr = tid >> 3;            // B: row k, 16 consecutive n
  const int bnc = (tid & 7) * 16;

  v8f acc[2][4];
#pragma unroll
  for (int mt = 0; mt < 2; ++mt)
#pragma unroll
    for (int nt = 0; nt < 4; ++nt) acc[mt][nt] = (v8f){0,0,0,0,0,0,0,0};

  // ---- prologue: stage tile k=0 into buffer 0
  {
    v16h aN;
    if (AHALF) {
      const half_t* A = (const half_t*)Aq;
      aN = *(const v16h*)(A + (size_t)(row0 + ar) * K + ac);
    } else {
      const float* A = (const float*)Aq;
      const float* ap = A + (size_t)(row0 + ar) * K + ac;
#pragma unroll
      for (int i = 0; i < 4; ++i) {
        float4 t = ((const float4*)ap)[i];
        aN[4*i+0] = (half_t)t.x; aN[4*i+1] = (half_t)t.y;
        aN[4*i+2] = (half_t)t.z; aN[4*i+3] = (half_t)t.w;
      }
    }
    v16h bN = *(const v16h*)(B + (size_t)bkr * N + col0 + bnc);
    *(v16h*)&As[0][ar][ac] = aN;
#pragma unroll
    for (int i = 0; i < 16; ++i) Bs[0][bnc + i][bkr] = bN[i];
  }
  __syncthreads();

  int cur = 0;
  for (int k0 = 0; k0 < K; k0 += 32) {
    const int kn = k0 + 32;
    const bool hn = kn < K;

    // ---- issue next tile's global loads early (latency overlap)
    v16h aN, bN;
    if (hn) {
      if (AHALF) {
        const half_t* A = (const half_t*)Aq;
        aN = *(const v16h*)(A + (size_t)(row0 + ar) * K + kn + ac);
      } else {
        const float* A = (const float*)Aq;
        const float* ap = A + (size_t)(row0 + ar) * K + kn + ac;
#pragma unroll
        for (int i = 0; i < 4; ++i) {
          float4 t = ((const float4*)ap)[i];
          aN[4*i+0] = (half_t)t.x; aN[4*i+1] = (half_t)t.y;
          aN[4*i+2] = (half_t)t.z; aN[4*i+3] = (half_t)t.w;
        }
      }
      bN = *(const v16h*)(B + (size_t)kn * N + col0 + bnc);
    }

    // ---- load ALL fragments first, then back-to-back WMMAs
    v16h afrag[2], bfrag[4];
#pragma unroll
    for (int mt = 0; mt < 2; ++mt) {
      int rr = wm + mt * 16 + ln;
      union { v16h v; h8 h[2]; } u;
      u.v = *(const v16h*)&As[cur][rr][hi * 16];
      afrag[mt] = u.v;
    }
#pragma unroll
    for (int nt = 0; nt < 4; ++nt)
      bfrag[nt] = *(const v16h*)&Bs[cur][wn + nt * 16 + ln][hi * 16];
#pragma unroll
    for (int nt = 0; nt < 4; ++nt)
#pragma unroll
      for (int mt = 0; mt < 2; ++mt)
        acc[mt][nt] = __builtin_amdgcn_wmma_f32_16x16x32_f16(
            false, afrag[mt], false, bfrag[nt], (short)0, acc[mt][nt],
            false, false);

    // ---- stage next tile into alternate buffer, single barrier per step
    if (hn) {
      int nxt = cur ^ 1;
      *(v16h*)&As[nxt][ar][ac] = aN;
#pragma unroll
      for (int i = 0; i < 16; ++i) Bs[nxt][bnc + i][bkr] = bN[i];
      __syncthreads();
      cur = nxt;
    }
  }

  // ---- epilogue
  if (EPI == 0) {
    // 384 = 3*128: a 128-wide tile never crosses a q/k/v boundary
    const int g = col0 / CDIM;                 // 0=q 1=k 2=v, block-uniform
    half_t* dstbuf = (g == 0) ? qb : (g == 1) ? kb : vb;
    const float smul = (g == 0) ? scale : 1.0f;
    const int colbase = col0 - g * CDIM;       // 0..256, within one group
#pragma unroll
    for (int mt = 0; mt < 2; ++mt) {
#pragma unroll
      for (int r = 0; r < 8; ++r) {
        int grow = row0 + wm + mt * 16 + hi * 8 + r;
        int bwin = grow / NTOK;
        int ntk  = grow - bwin * NTOK;
#pragma unroll
        for (int nt = 0; nt < 4; ++nt) {
          int within = colbase + wn + nt * 16 + ln;
          int hh = within >> 5;
          int dd = within & 31;
          float val = (acc[mt][nt][r] + bias[col0 + wn + nt * 16 + ln]) * smul;
          dstbuf[(((size_t)bwin * HEADS + hh) * NTOK + ntk) * HD + dd] =
              (half_t)val;
        }
      }
    }
  } else {
#pragma unroll
    for (int mt = 0; mt < 2; ++mt) {
#pragma unroll
      for (int r = 0; r < 8; ++r) {
        int grow = row0 + wm + mt * 16 + hi * 8 + r;
#pragma unroll
        for (int nt = 0; nt < 4; ++nt) {
          int gc = col0 + wn + nt * 16 + ln;
          outf[(size_t)grow * N + gc] = acc[mt][nt][r] + bias[gc];
        }
      }
    }
  }
}

// ---------------------------------------------------------------- attention
// One block = one (window, head). 128 threads = 4 waves, wave w owns 16 rows.
__global__ __launch_bounds__(128)
void wa_attn_wmma(const half_t* __restrict__ qb, const half_t* __restrict__ kb,
                  const half_t* __restrict__ vb, const float* __restrict__ bm,
                  half_t* __restrict__ ao) {
  __shared__ alignas(32) half_t Qs[64][32];   // zero-padded rows 49..63
  __shared__ alignas(32) half_t Ks[64][32];
  __shared__ alignas(32) half_t Vt[32][64];   // transposed: Vt[d][k]
  __shared__ alignas(32) half_t Ps[64][64];   // softmax probs, f16

  const int tid  = threadIdx.x;
  const int lane = tid & 31;
  const int wv   = tid >> 5;      // M-tile 0..3
  const int hi   = lane >> 4;
  const int ln   = lane & 15;
  const int hi8  = hi * 8;

  const int bh   = blockIdx.x;
  const int bwin = bh / HEADS;
  const int h    = bh - bwin * HEADS;
  const half_t* qp = qb + (size_t)bh * NTOK * HD;
  const half_t* kp = kb + (size_t)bh * NTOK * HD;
  const half_t* vp = vb + (size_t)bh * NTOK * HD;
  const float*  bmh = bm + (size_t)h * 64 * 64;

  // ---- stage Q,K (row layout) and V (transposed) into LDS, zero-pad
  {
    const uint4 z4 = make_uint4(0u, 0u, 0u, 0u);
#pragma unroll
    for (int it = 0; it < 2; ++it) {
      int q4  = tid + it * 128;        // uint4 index; 4 per 32-half row
      int row = q4 >> 2;
      uint4 dq = (row < NTOK) ? ((const uint4*)qp)[q4] : z4;
      ((uint4*)&Qs[0][0])[q4] = dq;
      uint4 dk = (row < NTOK) ? ((const uint4*)kp)[q4] : z4;
      ((uint4*)&Ks[0][0])[q4] = dk;
      uint4 dv = (row < NTOK) ? ((const uint4*)vp)[q4] : z4;
      int d0 = (q4 & 3) * 8;
      const half_t* hv = (const half_t*)&dv;
#pragma unroll
      for (int i = 0; i < 8; ++i) Vt[d0 + i][row] = hv[i];
    }
  }
  __syncthreads();

  // ---- S = Q K^T : preload 1 A frag + 4 B frags, then 4 WMMAs back-to-back
  v8f S[4];
#pragma unroll
  for (int nt = 0; nt < 4; ++nt) S[nt] = (v8f){0,0,0,0,0,0,0,0};
  {
    union { v16h v; h8 hh[2]; } aq;
    int rq = wv * 16 + ln;
    aq.v = *(const v16h*)&Qs[rq][hi * 16];
    v16h bkf[4];
#pragma unroll
    for (int nt = 0; nt < 4; ++nt)
      bkf[nt] = *(const v16h*)&Ks[nt * 16 + ln][hi * 16];
#pragma unroll
    for (int nt = 0; nt < 4; ++nt)
      S[nt] = __builtin_amdgcn_wmma_f32_16x16x32_f16(
          false, aq.v, false, bkf[nt], (short)0, S[nt], false, false);
  }

  // ---- add precomputed bias+mask (padding baked in), branchless
#pragma unroll
  for (int nt = 0; nt < 4; ++nt) {
    int j = nt * 16 + ln;
#pragma unroll
    for (int r = 0; r < 8; ++r) {
      int i = wv * 16 + hi * 8 + r;
      S[nt][r] += bmh[i * 64 + j];
    }
  }

  // ---- row softmax (row lives in one 16-lane group x 4 accums) -> Ps (f16)
#pragma unroll
  for (int r = 0; r < 8; ++r) {
    float m = S[0][r];
#pragma unroll
    for (int nt = 1; nt < 4; ++nt) m = fmaxf(m, S[nt][r]);
#pragma unroll
    for (int off = 1; off < 16; off <<= 1) m = fmaxf(m, __shfl_xor(m, off, 32));
    float sum = 0.f;
#pragma unroll
    for (int nt = 0; nt < 4; ++nt) {
      float e = __expf(S[nt][r] - m);
      S[nt][r] = e;
      sum += e;
    }
#pragma unroll
    for (int off = 1; off < 16; off <<= 1) sum += __shfl_xor(sum, off, 32);
    float inv = 1.f / sum;
    int i = wv * 16 + hi * 8 + r;
#pragma unroll
    for (int nt = 0; nt < 4; ++nt)
      Ps[i][nt * 16 + ln] = (half_t)(S[nt][r] * inv);
  }
  __syncthreads();

  // ---- O = P V : preload 2 A frags + 4 B frags, then 4 WMMAs back-to-back
  v8f O[2];
  O[0] = (v8f){0,0,0,0,0,0,0,0};
  O[1] = (v8f){0,0,0,0,0,0,0,0};
  {
    union { v16h v; h8 hh[2]; } ap[2];
    v16h bvf[2][2];
    int rp = wv * 16 + ln;
#pragma unroll
    for (int ks = 0; ks < 2; ++ks) {
      ap[ks].v = *(const v16h*)&Ps[rp][ks * 32 + hi * 16];
#pragma unroll
      for (int nt = 0; nt < 2; ++nt)
        bvf[ks][nt] = *(const v16h*)&Vt[nt * 16 + ln][ks * 32 + hi * 16];
    }
#pragma unroll
    for (int ks = 0; ks < 2; ++ks)
#pragma unroll
      for (int nt = 0; nt < 2; ++nt)
        O[nt] = __builtin_amdgcn_wmma_f32_16x16x32_f16(
            false, ap[ks].v, false, bvf[ks][nt], (short)0, O[nt],
            false, false);
  }

  // ---- store attention output f16 [B, 49, 384] (head-interleaved)
#pragma unroll
  for (int r = 0; r < 8; ++r) {
    int i = wv * 16 + hi * 8 + r;
    if (i < NTOK) {
#pragma unroll
      for (int nt = 0; nt < 2; ++nt) {
        int d = nt * 16 + ln;
        ao[((size_t)bwin * NTOK + i) * CDIM + h * HD + d] = (half_t)O[nt][r];
      }
    }
  }
}

// ---------------------------------------------------------------- launcher
extern "C" void kernel_launch(void* const* d_in, const int* in_sizes, int n_in,
                              void* d_out, int out_size, void* d_ws,
                              size_t ws_size, hipStream_t stream) {
  const float* x          = (const float*)d_in[0];
  const float* mask       = (const float*)d_in[1];
  const float* qkv_w      = (const float*)d_in[2];
  const float* qkv_b      = (const float*)d_in[3];
  const float* proj_w     = (const float*)d_in[4];
  const float* proj_b     = (const float*)d_in[5];
  const float* bias_table = (const float*)d_in[6];
  const int*   rel_index  = (const int*)d_in[7];
  float* out = (float*)d_out;

  const int Bwin = in_sizes[0] / (NTOK * CDIM);   // 4096
  const int M = Bwin * NTOK;                      // 200704 (divisible by 128)

  char* ws = (char*)d_ws;
  size_t off = 0;
  auto take = [&](size_t bytes) -> void* {
    void* p = ws + off;
    off += (bytes + 255) & ~(size_t)255;
    return p;
  };
  half_t* qkv_w_h  = (half_t*)take((size_t)CDIM * 3 * CDIM * sizeof(half_t));
  half_t* proj_w_h = (half_t*)take((size_t)CDIM * CDIM * sizeof(half_t));
  float*  bm       = (float*)take((size_t)HEADS * 64 * 64 * sizeof(float));
  size_t qkvN = (size_t)Bwin * HEADS * NTOK * HD;
  half_t* qb  = (half_t*)take(qkvN * sizeof(half_t));
  half_t* kb  = (half_t*)take(qkvN * sizeof(half_t));
  half_t* vb  = (half_t*)take(qkvN * sizeof(half_t));
  half_t* aoh = (half_t*)take((size_t)M * CDIM * sizeof(half_t));

  // 1) per-launch precompute: weight conversion + fused bias/mask table
  {
    int n1 = CDIM * 3 * CDIM;
    wa_f32_to_f16<<<(n1 + 255) / 256, 256, 0, stream>>>(qkv_w, qkv_w_h, n1);
    int n2 = CDIM * CDIM;
    wa_f32_to_f16<<<(n2 + 255) / 256, 256, 0, stream>>>(proj_w, proj_w_h, n2);
    int n3 = HEADS * 64 * 64;
    wa_bm_precompute<<<(n3 + 255) / 256, 256, 0, stream>>>(mask, bias_table,
                                                           rel_index, bm);
  }

  // 2) QKV GEMM -> q (scaled), k, v  f16 [B,H,49,32]
  const float scale = 0.17677669529663687f;   // 32^-0.5
  dim3 g1(3 * CDIM / 128, M / 128);           // (9, 1568)
  wa_gemm_wmma<false, 0><<<g1, 256, 0, stream>>>(
      (const void*)x, qkv_w_h, qkv_b, M, 3 * CDIM, CDIM,
      qb, kb, vb, nullptr, scale);

  // 3) fused window attention per (window, head)
  wa_attn_wmma<<<Bwin * HEADS, 128, 0, stream>>>(qb, kb, vb, bm, aoh);

  // 4) output projection + bias -> f32 d_out
  dim3 g2(CDIM / 128, M / 128);               // (3, 1568)
  wa_gemm_wmma<true, 1><<<g2, 256, 0, stream>>>(
      (const void*)aoh, proj_w_h, proj_b, M, CDIM, CDIM,
      nullptr, nullptr, nullptr, out, 1.0f);
}
